// LSTMIDS_2491081032411
// MI455X (gfx1250) — compile-verified, matrix operand order checked by reading, not run on hardware
//
#include <hip/hip_runtime.h>
#include <hip/hip_bf16.h>

typedef _Float16 v16h __attribute__((ext_vector_type(16)));
typedef float    v8f  __attribute__((ext_vector_type(8)));

#define TSEQ 256
#define IND  10
#define HD   20
#define GD   80
#define WPB  2          // waves per block
#define PF   6          // prefetch distance (timesteps)

// workspace byte offsets
#define WS_ABS   0      // u32 abs-max bits of x
#define WS_SX    16     // float: x quant scale
#define WS_WC    64     // _Float16[80*32] packed [W_hh | W_ih | 0] quantized
#define WS_W1Q   5632   // float[64*20]
#define WS_W2Q   10752  // float[32*64]
#define WS_W3Q   18944  // float[5*32]

// wave32 max-reduce via DS_SWIZZLE_B32 group-of-32 XOR mode:
// offset[14:10]=xor_mask, offset[9:5]=or_mask(0), offset[4:0]=and_mask(0x1f)
__device__ __forceinline__ float wmaxf(float v) {
  v = fmaxf(v, __int_as_float(__builtin_amdgcn_ds_swizzle(__float_as_int(v), 0x401f))); // xor16
  v = fmaxf(v, __int_as_float(__builtin_amdgcn_ds_swizzle(__float_as_int(v), 0x201f))); // xor8
  v = fmaxf(v, __int_as_float(__builtin_amdgcn_ds_swizzle(__float_as_int(v), 0x101f))); // xor4
  v = fmaxf(v, __int_as_float(__builtin_amdgcn_ds_swizzle(__float_as_int(v), 0x081f))); // xor2
  v = fmaxf(v, __int_as_float(__builtin_amdgcn_ds_swizzle(__float_as_int(v), 0x041f))); // xor1
  return v;
}
__device__ __forceinline__ float fsig(float x) { return 1.f / (1.f + __expf(-x)); }
__device__ __forceinline__ float ftanh(float x) {
  float e = __expf(-2.f * x);
  return (1.f - e) / (1.f + e);
}
// round(clip(v/s, lo, hi)) * s   (STE quant, eval mode)
__device__ __forceinline__ float qclip(float v, float s, float lo, float hi) {
  return roundf(fminf(fmaxf(v / s, lo), hi)) * s;
}

// ---------------- pass 1: global abs-max of x -------------------------------
__global__ __launch_bounds__(256) void absmax_kernel(const float* __restrict__ x,
                                                     size_t n, unsigned* out) {
  float m = 0.f;
  for (size_t i = (size_t)blockIdx.x * blockDim.x + threadIdx.x; i < n;
       i += (size_t)gridDim.x * blockDim.x)
    m = fmaxf(m, fabsf(x[i]));
  m = wmaxf(m);
  __shared__ float red[8];
  if ((threadIdx.x & 31) == 0) red[threadIdx.x >> 5] = m;
  __syncthreads();
  if (threadIdx.x == 0) {
    float t = red[0];
    for (int i = 1; i < (int)(blockDim.x >> 5); i++) t = fmaxf(t, red[i]);
    atomicMax(out, __float_as_uint(t));  // bit-compare valid: values >= 0
  }
}

// ---------------- pass 2: quantize + pack weights ---------------------------
__global__ __launch_bounds__(256) void prep_kernel(
    const float* __restrict__ w_ih, const float* __restrict__ w_hh,
    const float* __restrict__ w1, const float* __restrict__ w2,
    const float* __restrict__ w3, unsigned char* ws) {
  __shared__ float red[256];
  const int tid = threadIdx.x;
  auto bmaxabs = [&](const float* p, int n) -> float {
    float m = 0.f;
    for (int i = tid; i < n; i += 256) m = fmaxf(m, fabsf(p[i]));
    red[tid] = m;
    __syncthreads();
    for (int s = 128; s > 0; s >>= 1) {
      if (tid < s) red[tid] = fmaxf(red[tid], red[tid + s]);
      __syncthreads();
    }
    float r = red[0];
    __syncthreads();
    return r;
  };

  const float sih = bmaxabs(w_ih, GD * IND) * (1.f / 127.f) + 1e-8f;
  const float shh = bmaxabs(w_hh, GD * HD) * (1.f / 127.f) + 1e-8f;
  _Float16* Wc = (_Float16*)(ws + WS_WC);
  // packed combined weight: row g (gate), col K: [0..19]=W_hh, [20..29]=W_ih, [30..31]=0
  for (int i = tid; i < GD * 32; i += 256) {
    int g = i >> 5, k = i & 31;
    float v = 0.f;
    if (k < HD)            v = qclip(w_hh[g * HD + k], shh, -128.f, 127.f);
    else if (k < HD + IND) v = qclip(w_ih[g * IND + (k - HD)], sih, -128.f, 127.f);
    Wc[i] = (_Float16)v;
  }
  const float s1 = bmaxabs(w1, 64 * HD) * (1.f / 127.f) + 1e-8f;
  float* w1q = (float*)(ws + WS_W1Q);
  for (int i = tid; i < 64 * HD; i += 256) w1q[i] = qclip(w1[i], s1, -128.f, 127.f);
  const float s2 = bmaxabs(w2, 32 * 64) * (1.f / 127.f) + 1e-8f;
  float* w2q = (float*)(ws + WS_W2Q);
  for (int i = tid; i < 32 * 64; i += 256) w2q[i] = qclip(w2[i], s2, -128.f, 127.f);
  const float s3 = bmaxabs(w3, 5 * 32) * (1.f / 127.f) + 1e-8f;
  float* w3q = (float*)(ws + WS_W3Q);
  for (int i = tid; i < 5 * 32; i += 256) w3q[i] = qclip(w3[i], s3, -128.f, 127.f);
  if (tid == 0) {
    unsigned bits = *(unsigned*)(ws + WS_ABS);
    *(float*)(ws + WS_SX) = __uint_as_float(bits) * (1.f / 127.f) + 1e-8f;
  }
}

// ---------------- pass 3: fully fused LSTM scan + MLP head ------------------
// One wave owns 16 batch rows for all 256 timesteps. Per step:
//   gates[16x80] = [h16x20 | x16x10 | 0] (K=32)  x  Wc^T  + bias
// done as 5x V_WMMA_F32_16X16X32_F16 with weights resident in VGPRs.
// x loads are software-pipelined one full step ahead (double-buffered regs,
// manual 2x unroll so the buffer parity is static) + global_prefetch_b8
// issued PF steps ahead to stage lines in WGP$/L2.
__global__ __launch_bounds__(32 * WPB) void lstm_fused(
    const float* __restrict__ x, const float* __restrict__ b_ih,
    const float* __restrict__ b_hh, const float* __restrict__ b1,
    const float* __restrict__ b2, const float* __restrict__ b3,
    const _Float16* __restrict__ Wc, const float* __restrict__ sxp,
    const float* __restrict__ w1q, const float* __restrict__ w2q,
    const float* __restrict__ w3q, float* __restrict__ out, int B) {
  const int lane = threadIdx.x & 31;
  const int warp = threadIdx.x >> 5;
  const int tile = blockIdx.x * WPB + warp;  // 16 batch rows per wave
  if (tile * 16 >= B) return;                // wave-uniform exit
  const int row = lane & 15;                 // matrix row (M) this lane feeds
  const int bglobal = tile * 16 + row;
  const float sx = *sxp;

  __shared__ _Float16 sh_h[WPB][16][HD];  // h state (8-bit-quantized values, f16)
  __shared__ float    sh_c[WPB][16][HD];  // c state
  __shared__ float    sh_g[WPB][16][GD];  // activated gates / head scratch

  // state-update ownership: lane owns row (lane&15), j-window jb..jb+9
  // (div/mod-free: all indices are jb + compile-time q)
  const int jb = (lane < 16) ? 0 : 10;

  // zero state: 10 elements per lane
#pragma unroll
  for (int q = 0; q < 10; q++) {
    sh_h[warp][row][jb + q] = (_Float16)0.f;
    sh_c[warp][row][jb + q] = 0.f;
  }

  // B fragments (5 gate tiles of 16 cols) + combined bias, resident in VGPRs.
  // 16-bit 32x16 B layout: lanes 0-15 -> K {0..7,16..23}; lanes 16-31 -> +8.
  v16h Bf[5];
  float bcv[5];
  const int kbase = (lane >> 4) << 3;
#pragma unroll
  for (int nt = 0; nt < 5; nt++) {
    int g = nt * 16 + row;
    const _Float16* wrow = Wc + g * 32 + kbase;
#pragma unroll
    for (int e = 0; e < 8; e++) {
      Bf[nt][e]     = wrow[e];       // K = kbase + e
      Bf[nt][8 + e] = wrow[16 + e];  // K = kbase + 16 + e
    }
    bcv[nt] = b_ih[g] + b_hh[g];
  }

  const float* xrow = x + (size_t)bglobal * TSEQ * IND;
  // half-wave feature split: lanes 0-15 carry feats 0..3, lanes 16-31 feats 4..9
  const int fbase = (lane < 16) ? 0 : 4;
  const int fcnt  = (lane < 16) ? 4 : 6;

  // raw (unquantized) x loads: issue early, consume a full step later
  auto load_x = [&](int t, float (&dst)[6]) {
#pragma unroll
    for (int f = 0; f < 6; f++)
      dst[f] = (f < fcnt) ? xrow[t * IND + fbase + f] : 0.f;
  };

  // one LSTM timestep given pre-loaded raw x features
  auto step = [&](int t, const float (&xr)[6]) {
    _Float16 xq[6];
#pragma unroll
    for (int f = 0; f < 6; f++) xq[f] = (_Float16)qclip(xr[f], sx, -128.f, 127.f);
    asm volatile("s_wait_dscnt 0" ::: "memory");  // prior-step h stores (same wave)

    // --- A fragment: row = lane&15, K = [h(0..19) | x(20..29) | 0,0] ---
    v16h a;
    if (lane < 16) {
#pragma unroll
      for (int e = 0; e < 8; e++) a[e] = sh_h[warp][row][e];  // K 0..7
      a[8]  = sh_h[warp][row][16];
      a[9]  = sh_h[warp][row][17];
      a[10] = sh_h[warp][row][18];
      a[11] = sh_h[warp][row][19];                            // K 16..19
      a[12] = xq[0]; a[13] = xq[1]; a[14] = xq[2]; a[15] = xq[3];  // K 20..23
    } else {
#pragma unroll
      for (int e = 0; e < 8; e++) a[e] = sh_h[warp][row][8 + e];   // K 8..15
#pragma unroll
      for (int e = 0; e < 6; e++) a[8 + e] = xq[e];                // K 24..29
      a[14] = (_Float16)0.f;
      a[15] = (_Float16)0.f;                                       // K 30..31
    }

    // --- 5 WMMAs: gates = A x B + bias ---
    v8f acc[5];
#pragma unroll
    for (int nt = 0; nt < 5; nt++) {
#pragma unroll
      for (int e = 0; e < 8; e++) acc[nt][e] = bcv[nt];
      acc[nt] = __builtin_amdgcn_wmma_f32_16x16x32_f16(
          false, a, false, Bf[nt], (short)0, acc[nt], false, false);
    }

    // --- 6-bit gate quant (scale approximated per 16-row tile) ---
    float gm = 0.f;
#pragma unroll
    for (int nt = 0; nt < 5; nt++)
#pragma unroll
      for (int e = 0; e < 8; e++) gm = fmaxf(gm, fabsf(acc[nt][e]));
    const float gscale = wmaxf(gm) * (1.f / 31.f) + 1e-8f;

    // quantize + activation; scatter to LDS [m][g] (C/D layout: lane->N, vgpr->M)
#pragma unroll
    for (int nt = 0; nt < 5; nt++) {
      const int g = nt * 16 + row;
      const bool is_tanh = (g >= 2 * HD) && (g < 3 * HD);  // g-gate
#pragma unroll
      for (int e = 0; e < 8; e++) {
        const int m = (lane < 16) ? e : (e + 8);
        float v = qclip(acc[nt][e], gscale, -32.f, 31.f);
        float av = is_tanh ? qclip(ftanh(v), 1.f / 31.f, -32.f, 31.f)
                           : qclip(fsig(v), 1.f / 63.f, 0.f, 63.f);
        sh_g[warp][m][g] = av;
      }
    }
    asm volatile("s_wait_dscnt 0" ::: "memory");

    // --- state update: 10 hidden units (row, jb+q) per lane ---
    float cn[10], oa[10], hp[10];
    float cmax = 0.f;
#pragma unroll
    for (int q = 0; q < 10; q++) {
      float gi = sh_g[warp][row][jb + q];
      float gf = sh_g[warp][row][HD + jb + q];
      float gg = sh_g[warp][row][2 * HD + jb + q];
      float go = sh_g[warp][row][3 * HD + jb + q];
      float cp = gf * sh_c[warp][row][jb + q] + gi * gg;
      cn[q] = cp;
      oa[q] = go;
      cmax = fmaxf(cmax, fabsf(cp));
    }
    const float cscale = wmaxf(cmax) * (1.f / 31.f) + 1e-8f;
    float hmax = 0.f;
#pragma unroll
    for (int q = 0; q < 10; q++) {
      float cq = qclip(cn[q], cscale, -32.f, 31.f);
      sh_c[warp][row][jb + q] = cq;
      float tq = qclip(ftanh(cq), 1.f / 31.f, -32.f, 31.f);
      hp[q] = oa[q] * tq;
      hmax = fmaxf(hmax, fabsf(hp[q]));
    }
    const float hscale = wmaxf(hmax) * (1.f / 127.f) + 1e-8f;
#pragma unroll
    for (int q = 0; q < 10; q++)
      sh_h[warp][row][jb + q] = (_Float16)qclip(hp[q], hscale, -128.f, 127.f);
  };

  // --- scan, 2x unrolled with double-buffered x registers ---
  float xa[6], xb[6];
  load_x(0, xa);
  for (int t = 0; t < TSEQ; t += 2) {
    // stage lines PF steps ahead (lowers to global_prefetch_b8, no counter)
    if (t + PF < TSEQ) __builtin_prefetch(xrow + (t + PF) * IND + fbase, 0, 1);
    load_x(t + 1, xb);          // in flight across step(t)
    step(t, xa);
    if (t + PF + 1 < TSEQ) __builtin_prefetch(xrow + (t + PF + 1) * IND + fbase, 0, 1);
    if (t + 2 < TSEQ) load_x(t + 2, xa);  // in flight across step(t+1)
    step(t + 1, xb);
  }
  asm volatile("s_wait_dscnt 0" ::: "memory");

  // ---------------- MLP head: lane pair (r, r+16) shares row r --------------
  float hv[HD];
  float m0 = 0.f;
#pragma unroll
  for (int j = 0; j < HD; j++) {
    float v = fmaxf((float)sh_h[warp][row][j], 0.f);
    hv[j] = v;
    m0 = fmaxf(m0, v);
  }
  const float s0 = wmaxf(m0) * (1.f / 63.f) + 1e-8f;
#pragma unroll
  for (int j = 0; j < HD; j++) hv[j] = qclip(hv[j], s0, 0.f, 63.f);

  // layer1 (20->64): each half-wave computes 32 neurons of its row
  const int nb1 = (lane < 16) ? 0 : 32;
  float r1[32];
  float m1 = 0.f;
#pragma unroll
  for (int n = 0; n < 32; n++) {
    float d = b1[nb1 + n];
    const float* wr = w1q + (nb1 + n) * HD;
#pragma unroll
    for (int j = 0; j < HD; j++) d += wr[j] * hv[j];
    d = fmaxf(d, 0.f);
    r1[n] = d;
    m1 = fmaxf(m1, d);
  }
  const float s1 = wmaxf(m1) * (1.f / 63.f) + 1e-8f;
#pragma unroll
  for (int n = 0; n < 32; n++) sh_g[warp][row][nb1 + n] = qclip(r1[n], s1, 0.f, 63.f);
  asm volatile("s_wait_dscnt 0" ::: "memory");

  // layer2 (64->32): each half-wave computes 16 neurons; a1 read from LDS
  const int nb2 = (lane < 16) ? 0 : 16;
  float r2[16];
  float m2 = 0.f;
#pragma unroll
  for (int n = 0; n < 16; n++) {
    float d = b2[nb2 + n];
    const float* wr = w2q + (nb2 + n) * 64;
    for (int j = 0; j < 64; j++) d += wr[j] * sh_g[warp][row][j];
    d = fmaxf(d, 0.f);
    r2[n] = d;
    m2 = fmaxf(m2, d);
  }
  const float s2 = wmaxf(m2) * (1.f / 63.f) + 1e-8f;
  // park quantized a2: cols 0..15 -> sh_g[row][64..79], cols 16..31 -> sh_c[row][0..15]
#pragma unroll
  for (int n = 0; n < 16; n++) {
    float v = qclip(r2[n], s2, 0.f, 63.f);
    if (lane < 16) sh_g[warp][row][64 + n] = v;
    else           sh_c[warp][row][n] = v;
  }
  asm volatile("s_wait_dscnt 0" ::: "memory");

  // layer3 (32->5), no quant on output; lanes<16 store
  if (lane < 16) {
#pragma unroll
    for (int n = 0; n < 5; n++) {
      float d = b3[n];
      const float* wr = w3q + n * 32;
      for (int j = 0; j < 16; j++) d += wr[j] * sh_g[warp][row][64 + j];
      for (int j = 0; j < 16; j++) d += wr[16 + j] * sh_c[warp][row][j];
      out[(size_t)bglobal * 5 + n] = d;
    }
  }
}

extern "C" void kernel_launch(void* const* d_in, const int* in_sizes, int n_in,
                              void* d_out, int out_size, void* d_ws, size_t ws_size,
                              hipStream_t stream) {
  const float* x    = (const float*)d_in[0];
  const float* w_ih = (const float*)d_in[1];
  const float* w_hh = (const float*)d_in[2];
  const float* b_ih = (const float*)d_in[3];
  const float* b_hh = (const float*)d_in[4];
  const float* w1   = (const float*)d_in[5];
  const float* b1   = (const float*)d_in[6];
  const float* w2   = (const float*)d_in[7];
  const float* b2   = (const float*)d_in[8];
  const float* w3   = (const float*)d_in[9];
  const float* b3   = (const float*)d_in[10];
  unsigned char* ws = (unsigned char*)d_ws;

  hipMemsetAsync(d_ws, 0, 64, stream);  // zero atomic abs-max slot (capture-safe)

  const size_t nx = (size_t)in_sizes[0];
  absmax_kernel<<<2048, 256, 0, stream>>>(x, nx, (unsigned*)ws);
  prep_kernel<<<1, 256, 0, stream>>>(w_ih, w_hh, w1, w2, w3, ws);

  const int B = out_size / 5;                 // 8192
  const int blocks = (B + 16 * WPB - 1) / (16 * WPB);  // 256 blocks of 2 waves
  lstm_fused<<<blocks, 32 * WPB, 0, stream>>>(
      x, b_ih, b_hh, b1, b2, b3, (const _Float16*)(ws + WS_WC),
      (const float*)(ws + WS_SX), (const float*)(ws + WS_W1Q),
      (const float*)(ws + WS_W2Q), (const float*)(ws + WS_W3Q), (float*)d_out, B);
}
